// RGCNGateEncoder_42571715838486
// MI455X (gfx1250) — matrix operand order repaired and checked by reading, not compiled, not run on hardware
//
#include <hip/hip_runtime.h>
#include <hip/hip_bf16.h>

// RGCN gated 2-layer encoder for MI455X (gfx1250, wave32, WMMA bf16).
// - Edge message GEMM fused with f32 atomic scatter (never materializes the
//   E x H message tensor; out/h (20.5MB) and x (20.5MB) are L2-resident).
// - Edge-gather of x[col] rows done by the Tensor Data Mover in gather mode
//   (16 x 16-bit row indices, 512 f32 per row -> contiguous LDS tile).
// - bf16 WMMA (16x16x32) with f32 accumulation for all GEMMs.

#define HID 512

typedef __attribute__((ext_vector_type(16))) __bf16 v16bf;
typedef __attribute__((ext_vector_type(8)))  float  v8f;
typedef unsigned int tdm_v4u __attribute__((ext_vector_type(4)));
typedef int          tdm_v4i __attribute__((ext_vector_type(4)));
typedef int          tdm_v8i __attribute__((ext_vector_type(8)));

union BFrag { v16bf v; unsigned short u[16]; };

__device__ __forceinline__ void global_fadd(float* p, float v) {
  // Pin the native L2 atomic (no CAS-loop fallback). No-return -> STOREcnt.
  asm volatile("global_atomic_add_f32 %0, %1, off" : : "v"(p), "v"(v) : "memory");
}

// ---------------------------------------------------------------------------
// wT[r][o][i] = sum_b att[r][b] * basis[b][i][o]   (bf16, transposed for WMMA B)
// ---------------------------------------------------------------------------
__global__ __launch_bounds__(256) void rgcn_prep_w(
    const float* __restrict__ att, const float* __restrict__ basis,
    __bf16* __restrict__ wT, int R, int NB)
{
  int idx = blockIdx.x * 256 + threadIdx.x;          // r*H*H + o*H + i
  if (idx >= R * HID * HID) return;
  int r = idx / (HID * HID);
  int rem = idx - r * HID * HID;
  int o = rem / HID;
  int i = rem - o * HID;
  float s = 0.f;
  for (int b = 0; b < NB; ++b)
    s += att[r * NB + b] * basis[((size_t)b * HID + i) * HID + o];
  wT[idx] = (__bf16)s;
}

// rootT[o][i] = root[i][o]  (bf16 transpose)
__global__ __launch_bounds__(256) void rgcn_prep_rt(
    const float* __restrict__ root, __bf16* __restrict__ rT)
{
  int idx = blockIdx.x * 256 + threadIdx.x;          // o*H + i
  if (idx >= HID * HID) return;
  int o = idx / HID, i = idx - o * HID;
  rT[idx] = (__bf16)root[(size_t)i * HID + o];
}

__global__ __launch_bounds__(256) void rgcn_relu(float* __restrict__ p, int n)
{
  int i = blockIdx.x * 256 + threadIdx.x;
  if (i < n) p[i] = fmaxf(p[i], 0.f);
}

// ---------------------------------------------------------------------------
// out[n0b+m][:] = feat[n0b+m][:] @ rootT^T + bias   (16 nodes x 512 cols/block)
// ---------------------------------------------------------------------------
__global__ __launch_bounds__(256) void rgcn_node(
    const float* __restrict__ feat, const __bf16* __restrict__ rootT,
    const float* __restrict__ bias, float* __restrict__ out, int Nn)
{
  __shared__ __align__(16) float  xj[16][HID];
  __shared__ __align__(32) __bf16 ag[16][HID];
  const int tid = threadIdx.x;
  const int n0b = blockIdx.x * 16;

  #pragma unroll
  for (int it = 0; it < 8; ++it) {                   // 2048 float4 loads
    int idx = tid + 256 * it;
    int e = idx >> 7, q = (idx & 127) << 2;
    int nd = n0b + e; if (nd >= Nn) nd = Nn - 1;
    *(float4*)&xj[e][q] = *(const float4*)&feat[(size_t)nd * HID + q];
  }
  __syncthreads();
  #pragma unroll
  for (int it = 0; it < 32; ++it) {
    int idx = tid + 256 * it;
    int e = idx >> 9, k = idx & (HID - 1);
    ag[e][k] = (__bf16)xj[e][k];
  }
  __syncthreads();

  const int wave = tid >> 5, lane = tid & 31, hl = lane >> 4, lr = lane & 15;
  v8f acc[4] = {};
  for (int k0 = 0; k0 < HID; k0 += 32) {
    BFrag A;
    *(uint4*)&A.u[0] = *(const uint4*)&ag[lr][k0 + hl * 8];
    *(uint4*)&A.u[8] = *(const uint4*)&ag[lr][k0 + 16 + hl * 8];
    #pragma unroll
    for (int t = 0; t < 4; ++t) {
      int n0 = (wave * 4 + t) * 16;
      BFrag B;
      const __bf16* p = rootT + (size_t)(n0 + lr) * HID + k0 + hl * 8;
      *(uint4*)&B.u[0] = *(const uint4*)p;
      *(uint4*)&B.u[8] = *(const uint4*)(p + 16);
      acc[t] = __builtin_amdgcn_wmma_f32_16x16x32_bf16(
          false, A.v, false, B.v, (short)0, acc[t], false, false);
    }
  }
  #pragma unroll
  for (int t = 0; t < 4; ++t) {
    int n0 = (wave * 4 + t) * 16;
    #pragma unroll
    for (int j = 0; j < 8; ++j) {
      int m = j + hl * 8;
      int nd = n0b + m;
      if (nd < Nn)
        out[(size_t)nd * HID + n0 + lr] = acc[t][j] + bias[n0 + lr];
    }
  }
}

// ---------------------------------------------------------------------------
// Fused edge kernel: TDM-gather x[col] -> LDS, gate, per-relation message GEMM
// (bf16 WMMA, f32 acc), atomic scatter-add into out[row].
// 16 edges x 512 cols per block (8 waves).
// ---------------------------------------------------------------------------
__global__ __launch_bounds__(256) void rgcn_edge(
    const float* __restrict__ feat, const int* __restrict__ row,
    const int* __restrict__ col, const int* __restrict__ etype,
    const __bf16* __restrict__ wT, const float* __restrict__ gw,
    float* __restrict__ out, int E, int Nn)
{
  __shared__ __align__(16) float  xj[16][HID];   // TDM destination tile (32KB)
  __shared__ __align__(32) __bf16 ag[16][HID];
  __shared__ int   srow[16], srel[16];
  __shared__ float sgate[16];

  const int tid = threadIdx.x;
  const int e0 = blockIdx.x * 16;

  if (tid < 16) {
    int e = e0 + tid;
    bool v = e < E;
    srow[tid] = v ? row[e] : -1;
    srel[tid] = v ? etype[e] : 0;
  }

  // --- TDM gather: 16 rows of feat (indices = col[e0..e0+15], 16-bit) -> xj
  if (tid < 32) {                                 // wave 0 issues the DMA
    unsigned p[8];
    #pragma unroll
    for (int j = 0; j < 8; ++j) {
      int a = (e0 + 2 * j     < E) ? col[e0 + 2 * j]     : 0;
      int b = (e0 + 2 * j + 1 < E) ? col[e0 + 2 * j + 1] : 0;
      p[j] = ((unsigned)__builtin_amdgcn_readfirstlane(a) & 0xFFFFu) |
             ((unsigned)__builtin_amdgcn_readfirstlane(b) << 16);
    }
    unsigned long long ga = (unsigned long long)(uintptr_t)feat;
    unsigned ldsoff = (unsigned)(uintptr_t)(&xj[0][0]);   // flat LDS addr low bits = LDS offset
    // D# group0: count=1 | gather_mode (16-bit idx) | lds_addr | global_addr | type=2
    tdm_v4u g0 = { 1u | (1u << 31),
                   ldsoff,
                   (unsigned)ga,
                   ((unsigned)(ga >> 32) & 0x01FFFFFFu) | (2u << 30) };
    // D# group1: data_size=4B; tensor_dim0=512; tensor_dim1=Nn; tile_dim0=512;
    //            tile_dim1=16 (#indices); tensor_dim0_stride=512
    tdm_v8i g1 = { (int)(2u << 16),
                   (int)(512u << 16),
                   (int)((unsigned)Nn << 16),
                   (int)((512u << 16) | ((unsigned)Nn >> 16)),
                   16,
                   512,
                   0, 0 };
    tdm_v4i g2 = { (int)p[0], (int)p[1], (int)p[2], (int)p[3] };
    tdm_v4i g3 = { (int)p[4], (int)p[5], (int)p[6], (int)p[7] };
#if __clang_major__ >= 23
    tdm_v8i g4 = {};
    __builtin_amdgcn_tensor_load_to_lds(g0, g1, g2, g3, g4, 0);
#else
    __builtin_amdgcn_tensor_load_to_lds(g0, g1, g2, g3, 0);
#endif
    __builtin_amdgcn_s_wait_tensorcnt(0);
  }
  __syncthreads();

  const int wave = tid >> 5, lane = tid & 31;
  {                                               // gate = sigmoid(x_j . gw[r])
    int e = wave * 2 + (lane >> 4);
    int sub = lane & 15;
    const float* g = gw + (size_t)srel[e] * HID;
    float s = 0.f;
    #pragma unroll 4
    for (int k = sub; k < HID; k += 16) s += xj[e][k] * g[k];
    #pragma unroll
    for (int off = 8; off; off >>= 1) s += __shfl_xor(s, off, 16);
    if (sub == 0) sgate[e] = (srow[e] >= 0) ? 1.f / (1.f + __expf(-s)) : 0.f;
  }
  __syncthreads();

  #pragma unroll
  for (int it = 0; it < 32; ++it) {               // gated bf16 A tile
    int idx = tid + 256 * it;
    int e = idx >> 9, k = idx & (HID - 1);
    ag[e][k] = (__bf16)(sgate[e] * xj[e][k]);
  }
  __syncthreads();

  const int hl = lane >> 4, lr = lane & 15;
  const int rel_m = srel[lr];                     // relation of this lane's edge row
  const __bf16* w0 = wT;
  const __bf16* w1 = wT + (size_t)HID * HID;
  __builtin_prefetch(w0 + (size_t)(wave * 64 + lr) * HID, 0, 3);
  __builtin_prefetch(w1 + (size_t)(wave * 64 + lr) * HID, 0, 3);

  v8f acc[4] = {};
  const v16bf vz = {};
  for (int k0 = 0; k0 < HID; k0 += 32) {
    BFrag A;
    *(uint4*)&A.u[0] = *(const uint4*)&ag[lr][k0 + hl * 8];
    *(uint4*)&A.u[8] = *(const uint4*)&ag[lr][k0 + 16 + hl * 8];
    v16bf a0 = (rel_m == 0) ? A.v : vz;           // per-row relation masking:
    v16bf a1 = (rel_m == 0) ? vz : A.v;           // msg = a0@w0 + a1@w1
    #pragma unroll
    for (int t = 0; t < 4; ++t) {
      int n0 = (wave * 4 + t) * 16;
      BFrag B0, B1;
      const __bf16* p0 = w0 + (size_t)(n0 + lr) * HID + k0 + hl * 8;
      const __bf16* p1 = w1 + (size_t)(n0 + lr) * HID + k0 + hl * 8;
      *(uint4*)&B0.u[0] = *(const uint4*)p0;
      *(uint4*)&B0.u[8] = *(const uint4*)(p0 + 16);
      *(uint4*)&B1.u[0] = *(const uint4*)p1;
      *(uint4*)&B1.u[8] = *(const uint4*)(p1 + 16);
      acc[t] = __builtin_amdgcn_wmma_f32_16x16x32_bf16(
          false, a0, false, B0.v, (short)0, acc[t], false, false);
      acc[t] = __builtin_amdgcn_wmma_f32_16x16x32_bf16(
          false, a1, false, B1.v, (short)0, acc[t], false, false);
    }
  }

  #pragma unroll
  for (int t = 0; t < 4; ++t) {                   // segment-sum via native f32 atomics
    int n0 = (wave * 4 + t) * 16;
    #pragma unroll
    for (int j = 0; j < 8; ++j) {
      int m = j + hl * 8;
      int dr = srow[m];
      if (dr >= 0)
        global_fadd(out + (size_t)dr * HID + n0 + lr, acc[t][j]);
    }
  }
}

// ---------------------------------------------------------------------------
extern "C" void kernel_launch(void* const* d_in, const int* in_sizes, int n_in,
                              void* d_out, int out_size, void* d_ws, size_t ws_size,
                              hipStream_t stream)
{
  const float* x      = (const float*)d_in[0];
  const int*   ei     = (const int*)  d_in[1];
  const int*   et     = (const int*)  d_in[2];
  const float* basis1 = (const float*)d_in[3];
  const float* att1   = (const float*)d_in[4];
  const float* gw1    = (const float*)d_in[5];
  const float* root1  = (const float*)d_in[6];
  const float* bias1  = (const float*)d_in[7];
  const float* basis2 = (const float*)d_in[8];
  const float* att2   = (const float*)d_in[9];
  const float* gw2    = (const float*)d_in[10];
  const float* root2  = (const float*)d_in[11];
  const float* bias2  = (const float*)d_in[12];
  float* out = (float*)d_out;

  const int Nn = in_sizes[0] / HID;           // 10000
  const int E  = in_sizes[1] / 2;             // 160000
  const int NB = in_sizes[3] / (HID * HID);   // 30
  const int R  = in_sizes[4] / NB;            // 2

  const int* rowp = ei;
  const int* colp = ei + E;

  // workspace layout (~23.6 MB): h | wT1 | rootT1 | wT2 | rootT2
  char* ws = (char*)d_ws;
  size_t hb = (((size_t)Nn * HID * 4) + 255) & ~(size_t)255;
  float*  h   = (float*)ws;
  __bf16* wT1 = (__bf16*)(ws + hb);
  __bf16* rT1 = wT1 + (size_t)R * HID * HID;
  __bf16* wT2 = rT1 + (size_t)HID * HID;
  __bf16* rT2 = wT2 + (size_t)R * HID * HID;

  dim3 blk(256);
  int gW  = (R * HID * HID + 255) / 256;
  int gRT = (HID * HID + 255) / 256;
  int gN  = (Nn + 15) / 16;
  int gE  = (E + 15) / 16;
  int gRe = (Nn * HID + 255) / 256;

  rgcn_prep_w <<<gW,  blk, 0, stream>>>(att1, basis1, wT1, R, NB);
  rgcn_prep_w <<<gW,  blk, 0, stream>>>(att2, basis2, wT2, R, NB);
  rgcn_prep_rt<<<gRT, blk, 0, stream>>>(root1, rT1);
  rgcn_prep_rt<<<gRT, blk, 0, stream>>>(root2, rT2);

  // layer 1: h = relu( scatter(msg) + x@root1 + bias1 )
  rgcn_node<<<gN, blk, 0, stream>>>(x, rT1, bias1, h, Nn);
  rgcn_edge<<<gE, blk, 0, stream>>>(x, rowp, colp, et, wT1, gw1, h, E, Nn);
  rgcn_relu<<<gRe, blk, 0, stream>>>(h, Nn * HID);

  // layer 2: out = scatter(msg) + h@root2 + bias2
  rgcn_node<<<gN, blk, 0, stream>>>(h, rT2, bias2, out, Nn);
  rgcn_edge<<<gE, blk, 0, stream>>>(h, rowp, colp, et, wT2, gw2, out, E, Nn);
}
